// get_model_59639915872405
// MI455X (gfx1250) — matrix operand order repaired and checked by reading, not compiled, not run on hardware
//
#include <hip/hip_runtime.h>
#include <hip/hip_bf16.h>

typedef __attribute__((ext_vector_type(16))) _Float16 v16h;
typedef __attribute__((ext_vector_type(8)))  _Float16 v8h;
typedef __attribute__((ext_vector_type(8)))  float    v8f;

// ---------------------------------------------------------------------------
// WMMA GEMM:  C(MxN) = A(MxK,f16,row-major) * Bt(NxK,f16,row-major)^T + bias
// K multiple of 32, M,N multiples of 32.  One wave -> one 32x32 C tile
// (2x2 register-blocked 16x16 WMMA tiles: 4 wmmas per 8 b128 loads).
// Operand fragments follow the CDNA5 16-bit A/B VGPR layout tables, giving
// contiguous 16B loads per lane.
// ---------------------------------------------------------------------------
__global__ void __launch_bounds__(256)
gemm_wmma_kernel(const _Float16* __restrict__ A,
                 const _Float16* __restrict__ Bt,
                 const float*    __restrict__ bias,
                 _Float16* __restrict__ C16,
                 float*    __restrict__ C32,
                 int M, int N, int K)
{
    const int lane  = threadIdx.x & 31;
    const int wave  = blockIdx.x * (blockDim.x >> 5) + (threadIdx.x >> 5);
    const int tilesN = N >> 5;                 // 32-wide tiles
    const int tilesM = M >> 5;
    if (wave >= tilesM * tilesN) return;       // uniform per wave: EXEC stays all-ones
    const int tm = wave / tilesN;
    const int tn = wave - tm * tilesN;

    const int l15 = lane & 15;
    const int h   = lane >> 4;
    const int m0  = tm * 32;
    const int n0  = tn * 32;

    // A: lane = row M (two 16-row tiles) ; B: lane = col N (two 16-col tiles)
    const _Float16* Arow0 = A  + (size_t)(m0 + l15) * K;
    const _Float16* Arow1 = A  + (size_t)(m0 + 16 + l15) * K;
    const _Float16* Brow0 = Bt + (size_t)(n0 + l15) * K;
    const _Float16* Brow1 = Bt + (size_t)(n0 + 16 + l15) * K;

    const float bv0 = bias ? bias[n0 + l15]      : 0.0f;
    const float bv1 = bias ? bias[n0 + 16 + l15] : 0.0f;
    v8f acc00 = { bv0, bv0, bv0, bv0, bv0, bv0, bv0, bv0 };
    v8f acc01 = { bv1, bv1, bv1, bv1, bv1, bv1, bv1, bv1 };
    v8f acc10 = acc00;
    v8f acc11 = acc01;

    for (int k0 = 0; k0 < K; k0 += 32) {
        // A 16x32 layout: lanes 0-15 K=0..7 / 16..23 ; lanes 16-31 K=8..15 / 24..31
        v8h a0lo = *(const v8h*)(Arow0 + k0 + 8 * h);
        v8h a0hi = *(const v8h*)(Arow0 + k0 + 16 + 8 * h);
        v8h a1lo = *(const v8h*)(Arow1 + k0 + 8 * h);
        v8h a1hi = *(const v8h*)(Arow1 + k0 + 16 + 8 * h);
        // B 32x16 layout: lanes 0-15 K=0..7 / 16..23 ; lanes 16-31 K=8..15 / 24..31
        v8h b0lo = *(const v8h*)(Brow0 + k0 + 8 * h);
        v8h b0hi = *(const v8h*)(Brow0 + k0 + 16 + 8 * h);
        v8h b1lo = *(const v8h*)(Brow1 + k0 + 8 * h);
        v8h b1hi = *(const v8h*)(Brow1 + k0 + 16 + 8 * h);

        v16h a0 = __builtin_shufflevector(a0lo, a0hi, 0,1,2,3,4,5,6,7,8,9,10,11,12,13,14,15);
        v16h a1 = __builtin_shufflevector(a1lo, a1hi, 0,1,2,3,4,5,6,7,8,9,10,11,12,13,14,15);
        v16h b0 = __builtin_shufflevector(b0lo, b0hi, 0,1,2,3,4,5,6,7,8,9,10,11,12,13,14,15);
        v16h b1 = __builtin_shufflevector(b1lo, b1hi, 0,1,2,3,4,5,6,7,8,9,10,11,12,13,14,15);

        __builtin_prefetch(Arow0 + k0 + 128, 0, 1);   // global_prefetch_b8
        __builtin_prefetch(Arow1 + k0 + 128, 0, 1);
        __builtin_prefetch(Brow0 + k0 + 128, 0, 1);
        __builtin_prefetch(Brow1 + k0 + 128, 0, 1);

        acc00 = __builtin_amdgcn_wmma_f32_16x16x32_f16(false, a0, false, b0, (short)0, acc00, false, false);
        acc01 = __builtin_amdgcn_wmma_f32_16x16x32_f16(false, a0, false, b1, (short)0, acc01, false, false);
        acc10 = __builtin_amdgcn_wmma_f32_16x16x32_f16(false, a1, false, b0, (short)0, acc10, false, false);
        acc11 = __builtin_amdgcn_wmma_f32_16x16x32_f16(false, a1, false, b1, (short)0, acc11, false, false);
    }

    // C/D layout: VGPR r -> row r + 8h within a 16-row tile; lane%16 = column
    const int c0 = n0 + l15;
    const int c1 = n0 + 16 + l15;
    #pragma unroll
    for (int r = 0; r < 8; ++r) {
        size_t r0 = (size_t)(m0 + r + 8 * h) * N;
        size_t r1 = (size_t)(m0 + 16 + r + 8 * h) * N;
        if (C32) {
            C32[r0 + c0] = acc00[r];
            C32[r0 + c1] = acc01[r];
            C32[r1 + c0] = acc10[r];
            C32[r1 + c1] = acc11[r];
        } else {
            C16[r0 + c0] = (_Float16)acc00[r];
            C16[r0 + c1] = (_Float16)acc01[r];
            C16[r1 + c0] = (_Float16)acc10[r];
            C16[r1 + c1] = (_Float16)acc11[r];
        }
    }
}

// ---------------------------------------------------------------------------
// Weight prep: Wt[n*Kpad + k] = (k < Kin) ? W[k*N + n] : 0   (f32 -> f16)
// ---------------------------------------------------------------------------
__global__ void wt_kernel(const float* __restrict__ W, _Float16* __restrict__ Wt,
                          int Kin, int Kpad, int N, size_t total)
{
    size_t e = (size_t)blockIdx.x * blockDim.x + threadIdx.x;
    if (e >= total) return;
    int n = (int)(e / Kpad);
    int k = (int)(e - (size_t)n * Kpad);
    Wt[e] = (k < Kin) ? (_Float16)W[(size_t)k * N + n] : (_Float16)0.0f;
}

// ---------------------------------------------------------------------------
// Split (B,6,N) -> pts (B,N,3), norm (B,N,3)
// ---------------------------------------------------------------------------
__global__ void split_kernel(const float* __restrict__ xyz,
                             float* __restrict__ pts, float* __restrict__ nrm,
                             int N, size_t total)
{
    size_t e = (size_t)blockIdx.x * blockDim.x + threadIdx.x;
    if (e >= total) return;
    int b = (int)(e / N);
    int n = (int)(e - (size_t)b * N);
    const float* src = xyz + (size_t)b * 6 * N;
    float* p = pts + e * 3;
    float* q = nrm + e * 3;
    p[0] = src[0 * N + n]; p[1] = src[1 * N + n]; p[2] = src[2 * N + n];
    q[0] = src[3 * N + n]; q[1] = src[4 * N + n]; q[2] = src[5 * N + n];
}

// ---------------------------------------------------------------------------
// Farthest point sampling: one block per batch. dists live in LDS.
// Emits `far` BEFORE updating (matches the reference scan carry).
// ---------------------------------------------------------------------------
__global__ void fps_kernel(const float* __restrict__ xyz, int* __restrict__ out,
                           int N, int npoint)
{
    __shared__ float dist[4096];
    __shared__ float rv[256];
    __shared__ int   ri[256];
    __shared__ int   sfar;
    const int b = blockIdx.x, tid = threadIdx.x;
    const float* P = xyz + (size_t)b * N * 3;

    for (int i = tid; i < N; i += 256) dist[i] = 1e10f;
    if (tid == 0) sfar = 0;
    __syncthreads();

    for (int t = 0; t < npoint; ++t) {
        int far = sfar;
        if (tid == 0) out[b * npoint + t] = far;
        float fx = P[far * 3 + 0], fy = P[far * 3 + 1], fz = P[far * 3 + 2];
        float bv = -1.0f; int bi = 0;
        for (int i = tid; i < N; i += 256) {
            float dx = P[i * 3 + 0] - fx;
            float dy = P[i * 3 + 1] - fy;
            float dz = P[i * 3 + 2] - fz;
            float d  = dx * dx + dy * dy + dz * dz;
            float nd = fminf(dist[i], d);
            dist[i] = nd;
            if (nd > bv) { bv = nd; bi = i; }   // first-max within ascending scan
        }
        rv[tid] = bv; ri[tid] = bi;
        __syncthreads();
        for (int s = 128; s > 0; s >>= 1) {
            if (tid < s) {
                if (rv[tid + s] > rv[tid] ||
                    (rv[tid + s] == rv[tid] && ri[tid + s] < ri[tid])) {
                    rv[tid] = rv[tid + s]; ri[tid] = ri[tid + s];
                }
            }
            __syncthreads();
        }
        if (tid == 0) sfar = ri[0];
        __syncthreads();
    }
}

// gather new_xyz[b,s,:] = src[b, fidx[b,s], :]
__global__ void gather_kernel(const float* __restrict__ src, const int* __restrict__ fidx,
                              float* __restrict__ dst, int N, size_t total)
{
    size_t e = (size_t)blockIdx.x * blockDim.x + threadIdx.x;
    if (e >= total) return;                      // total = B*S
    int b = (int)(e / (total / 32));             // S = total/32
    int j = fidx[e];
    const float* p = src + ((size_t)b * N + j) * 3;
    float* q = dst + e * 3;
    q[0] = p[0]; q[1] = p[1]; q[2] = p[2];
}

// ---------------------------------------------------------------------------
// Ball query: first `nsample` indices (ascending) with d^2 <= r^2, padded with
// the first hit (mirrors sort-then-replace in the reference).
// ---------------------------------------------------------------------------
__global__ void ballquery_kernel(const float* __restrict__ src, const float* __restrict__ q,
                                 int* __restrict__ gidx, int N, int S, int nsample,
                                 float r2, size_t total)
{
    size_t e = (size_t)blockIdx.x * blockDim.x + threadIdx.x;
    if (e >= total) return;                      // total = B*S
    int b = (int)(e / S);
    const float* Q = q + e * 3;
    float qx = Q[0], qy = Q[1], qz = Q[2];
    const float* P = src + (size_t)b * N * 3;
    int* out = gidx + e * nsample;
    int cnt = 0;
    for (int j = 0; j < N && cnt < nsample; ++j) {
        float dx = P[j * 3 + 0] - qx;
        float dy = P[j * 3 + 1] - qy;
        float dz = P[j * 3 + 2] - qz;
        if (dx * dx + dy * dy + dz * dz <= r2) out[cnt++] = j;
    }
    int first = (cnt > 0) ? out[0] : 0;
    for (int k = cnt; k < nsample; ++k) out[k] = first;
}

// ---------------------------------------------------------------------------
// Grouping: out row (b,s,k) gets [xyz[j]-new_xyz | pfeat[j] | 0-pad] as f16
// ---------------------------------------------------------------------------
__global__ void group_kernel(const float* __restrict__ xyz, const float* __restrict__ nxyz,
                             const float* __restrict__ pfeat, const int* __restrict__ gidx,
                             _Float16* __restrict__ out,
                             int N, int S, int K, int Cp, int Cpad, size_t total)
{
    size_t e = (size_t)blockIdx.x * blockDim.x + threadIdx.x;
    if (e >= total) return;                       // total = B*S*K*Cpad
    size_t row = e / Cpad;
    int c = (int)(e - row * Cpad);
    int s = (int)((row / K) % S);
    int b = (int)(row / ((size_t)S * K));
    int j = gidx[row];
    float v;
    if (c < 3)
        v = xyz[((size_t)b * N + j) * 3 + c] - nxyz[((size_t)b * S + s) * 3 + c];
    else if (c < 3 + Cp)
        v = pfeat[((size_t)b * N + j) * Cp + (c - 3)];
    else
        v = 0.0f;
    out[e] = (_Float16)v;
}

// group_all for SA3: [l2x | l2p | pad]
__global__ void groupall_kernel(const float* __restrict__ l2x, const float* __restrict__ l2p,
                                _Float16* __restrict__ out, int K, int Cp, int Cpad,
                                size_t total)
{
    size_t e = (size_t)blockIdx.x * blockDim.x + threadIdx.x;
    if (e >= total) return;                       // total = B*K*Cpad
    size_t row = e / Cpad;
    int c = (int)(e - row * Cpad);
    float v;
    if (c < 3)            v = l2x[row * 3 + c];
    else if (c < 3 + Cp)  v = l2p[row * Cp + (c - 3)];
    else                  v = 0.0f;
    out[e] = (_Float16)v;
}

// ---------------------------------------------------------------------------
// BatchNorm stats: one block per channel; biased variance, inv = rsqrt(var+eps)
// ---------------------------------------------------------------------------
__global__ void bnstats_kernel(const _Float16* __restrict__ Z,
                               float* __restrict__ mean, float* __restrict__ inv,
                               int M, int N)
{
    __shared__ float s1[256];
    __shared__ float s2[256];
    int c = blockIdx.x, tid = threadIdx.x;
    float a = 0.0f, b = 0.0f;
    for (int r = tid; r < M; r += 256) {
        float z = (float)Z[(size_t)r * N + c];
        a += z; b += z * z;
    }
    s1[tid] = a; s2[tid] = b;
    __syncthreads();
    for (int s = 128; s > 0; s >>= 1) {
        if (tid < s) { s1[tid] += s1[tid + s]; s2[tid] += s2[tid + s]; }
        __syncthreads();
    }
    if (tid == 0) {
        float mu  = s1[0] / (float)M;
        float var = fmaxf(s2[0] / (float)M - mu * mu, 0.0f);
        mean[c] = mu;
        inv[c]  = rsqrtf(var + 1e-5f);
    }
}

__global__ void bnrelu_kernel(const _Float16* __restrict__ Z,
                              const float* __restrict__ mean, const float* __restrict__ inv,
                              const float* __restrict__ gamma, const float* __restrict__ beta,
                              _Float16* __restrict__ Y, float* __restrict__ Y32,
                              size_t total, int N)
{
    size_t e = (size_t)blockIdx.x * blockDim.x + threadIdx.x;
    if (e >= total) return;
    int c = (int)(e % N);
    float z = (float)Z[e];
    float y = gamma[c] * (z - mean[c]) * inv[c] + beta[c];
    y = fmaxf(y, 0.0f);
    Y[e] = (_Float16)y;
    if (Y32) Y32[e] = y;
}

// max over K of Y[(bs*K+k)*C + c]
__global__ void maxpool_kernel(const _Float16* __restrict__ Y,
                               float* __restrict__ outF, _Float16* __restrict__ outH,
                               int K, int C, size_t total)
{
    size_t e = (size_t)blockIdx.x * blockDim.x + threadIdx.x;
    if (e >= total) return;                       // total = BS*C
    size_t bs = e / C;
    int c = (int)(e - bs * C);
    float m = -3.4e38f;
    for (int k = 0; k < K; ++k)
        m = fmaxf(m, (float)Y[((size_t)bs * K + k) * C + c]);
    if (outF) outF[e] = m;
    if (outH) outH[e] = (_Float16)m;
}

// ---------------------------------------------------------------------------
// Host orchestration
// ---------------------------------------------------------------------------
static inline unsigned gblk(size_t n) { return (unsigned)((n + 255) / 256); }

static void run_gemm(hipStream_t s, const _Float16* A, const _Float16* Bt, const float* bias,
                     _Float16* C16, float* C32, int M, int N, int K)
{
    int tiles  = (M / 32) * (N / 32);             // 32x32 tile per wave
    int blocks = (tiles + 7) / 8;                 // 8 waves / 256-thread block
    gemm_wmma_kernel<<<blocks, 256, 0, s>>>(A, Bt, bias, C16, C32, M, N, K);
}

static void run_bn(hipStream_t s, const _Float16* Z, const float* gamma, const float* beta,
                   float* mean, float* inv, _Float16* Y, float* Y32, int M, int N)
{
    bnstats_kernel<<<N, 256, 0, s>>>(Z, mean, inv, M, N);
    size_t tot = (size_t)M * N;
    bnrelu_kernel<<<gblk(tot), 256, 0, s>>>(Z, mean, inv, gamma, beta, Y, Y32, tot, N);
}

extern "C" void kernel_launch(void* const* d_in, const int* in_sizes, int n_in,
                              void* d_out, int out_size, void* d_ws, size_t ws_size,
                              hipStream_t stream)
{
    (void)out_size; (void)ws_size;
    const int B = 32, NPTS = 4096;

    // ---- input pointer resolution (handles dict-order vs sorted-pytree order) ----
    const float* xyz;
    int pbase;
    if (in_sizes[0] == B * 6 * NPTS) { xyz = (const float*)d_in[0];        pbase = 1; }
    else                             { xyz = (const float*)d_in[n_in - 1]; pbase = 0; }

    // canonical order: sa1(12), sa2(12), sa3(12), fc(16), out(2)
    const float* P[54];
    bool sortedOrder = (in_sizes[pbase] == 1024 * 512);   // fc[0].W first => sorted keys
    for (int i = 0; i < 54; ++i) {
        int idx = sortedOrder ? ((i < 36) ? 18 + i : i - 36) : i;
        P[i] = (const float*)d_in[pbase + idx];
    }

    // ---- workspace bump allocator ----
    char* w = (char*)d_ws;
    auto alloc = [&](size_t bytes) -> void* {
        void* p = (void*)w;
        w += (bytes + 255) & ~(size_t)255;
        return p;
    };

    float* pts   = (float*)alloc((size_t)B * NPTS * 3 * 4);
    float* nrm   = (float*)alloc((size_t)B * NPTS * 3 * 4);
    int*   fidx1 = (int*)  alloc((size_t)B * 512 * 4);
    float* l1x   = (float*)alloc((size_t)B * 512 * 3 * 4);
    int*   gidx1 = (int*)  alloc((size_t)B * 512 * 32 * 4);
    float* l1p   = (float*)alloc((size_t)B * 512 * 128 * 4);
    int*   fidx2 = (int*)  alloc((size_t)B * 128 * 4);
    float* l2x   = (float*)alloc((size_t)B * 128 * 3 * 4);
    int*   gidx2 = (int*)  alloc((size_t)B * 128 * 64 * 4);
    float* l2p   = (float*)alloc((size_t)B * 128 * 256 * 4);
    float* meanb = (float*)alloc(1024 * 4);
    float* invb  = (float*)alloc(1024 * 4);
    const size_t BIG = (size_t)524288 * 128;      // max M*C in halfs
    _Float16* buf0 = (_Float16*)alloc(BIG * 2);
    _Float16* buf1 = (_Float16*)alloc(BIG * 2);

    // ---- GEMM layer table: {param base, Kin, Kpad, N} ----
    struct L { int p, Kin, Kpad, N; };
    const L LY[14] = {
        { 0,    6,   32,    64}, { 4,   64,   64,    64}, { 8,   64,   64,   128},  // sa1
        {12,  131,  160,   128}, {16,  128,  128,   128}, {20,  128,  128,   256},  // sa2
        {24,  259,  288,   256}, {28,  256,  256,   512}, {32,  512,  512,  1024},  // sa3
        {36, 1024, 1024,   512}, {40,  512,  512,   256}, {44,  256,  256,   512},  // fc
        {48,  512,  512,  1024},
        {52, 1024, 1024, 12288},                                                    // out
    };
    _Float16* Wt[14];
    for (int i = 0; i < 14; ++i) {
        size_t tot = (size_t)LY[i].Kpad * LY[i].N;
        Wt[i] = (_Float16*)alloc(tot * 2);
        wt_kernel<<<gblk(tot), 256, 0, stream>>>(P[LY[i].p], Wt[i],
                                                 LY[i].Kin, LY[i].Kpad, LY[i].N, tot);
    }
    auto bias_of  = [&](int i) { return P[LY[i].p + 1]; };
    auto gamma_of = [&](int i) { return P[LY[i].p + 2]; };
    auto beta_of  = [&](int i) { return P[LY[i].p + 3]; };

    // ---- split input ----
    split_kernel<<<gblk((size_t)B * NPTS), 256, 0, stream>>>(xyz, pts, nrm, NPTS,
                                                             (size_t)B * NPTS);

    // ================= SA1: 4096 pts -> 512 centroids, K=32, C 6->64->64->128 ====
    fps_kernel<<<B, 256, 0, stream>>>(pts, fidx1, NPTS, 512);
    gather_kernel<<<gblk((size_t)B * 512), 256, 0, stream>>>(pts, fidx1, l1x, NPTS,
                                                             (size_t)B * 512);
    ballquery_kernel<<<gblk((size_t)B * 512), 256, 0, stream>>>(
        pts, l1x, gidx1, NPTS, 512, 32, 0.2f * 0.2f, (size_t)B * 512);
    {
        int M = B * 512 * 32;
        size_t tot = (size_t)M * 32;
        group_kernel<<<gblk(tot), 256, 0, stream>>>(pts, l1x, nrm, gidx1, buf0,
                                                    NPTS, 512, 32, 3, 32, tot);
        for (int i = 0; i < 3; ++i) {
            run_gemm(stream, buf0, Wt[i], bias_of(i), buf1, nullptr, M, LY[i].N, LY[i].Kpad);
            run_bn(stream, buf1, gamma_of(i), beta_of(i), meanb, invb, buf0, nullptr,
                   M, LY[i].N);
        }
        maxpool_kernel<<<gblk((size_t)B * 512 * 128), 256, 0, stream>>>(
            buf0, l1p, nullptr, 32, 128, (size_t)B * 512 * 128);
    }

    // ================= SA2: 512 -> 128 centroids, K=64, C 131->128->128->256 =====
    fps_kernel<<<B, 256, 0, stream>>>(l1x, fidx2, 512, 128);
    gather_kernel<<<gblk((size_t)B * 128), 256, 0, stream>>>(l1x, fidx2, l2x, 512,
                                                             (size_t)B * 128);
    ballquery_kernel<<<gblk((size_t)B * 128), 256, 0, stream>>>(
        l1x, l2x, gidx2, 512, 128, 64, 0.4f * 0.4f, (size_t)B * 128);
    {
        int M = B * 128 * 64;
        size_t tot = (size_t)M * 160;
        group_kernel<<<gblk(tot), 256, 0, stream>>>(l1x, l2x, l1p, gidx2, buf0,
                                                    512, 128, 64, 128, 160, tot);
        for (int i = 3; i < 6; ++i) {
            run_gemm(stream, buf0, Wt[i], bias_of(i), buf1, nullptr, M, LY[i].N, LY[i].Kpad);
            run_bn(stream, buf1, gamma_of(i), beta_of(i), meanb, invb, buf0, nullptr,
                   M, LY[i].N);
        }
        maxpool_kernel<<<gblk((size_t)B * 128 * 256), 256, 0, stream>>>(
            buf0, l2p, nullptr, 64, 256, (size_t)B * 128 * 256);
    }

    // ================= SA3: group_all over 128, C 259->256->512->1024 ============
    {
        int M = B * 128;
        size_t tot = (size_t)M * 288;
        groupall_kernel<<<gblk(tot), 256, 0, stream>>>(l2x, l2p, buf0, 128, 256, 288, tot);
        for (int i = 6; i < 9; ++i) {
            run_gemm(stream, buf0, Wt[i], bias_of(i), buf1, nullptr, M, LY[i].N, LY[i].Kpad);
            run_bn(stream, buf1, gamma_of(i), beta_of(i), meanb, invb, buf0, nullptr,
                   M, LY[i].N);
        }
        // max over K=128 -> (B,1024) as f16 FC input in buf1
        maxpool_kernel<<<gblk((size_t)B * 1024), 256, 0, stream>>>(
            buf0, nullptr, buf1, 128, 1024, (size_t)B * 1024);
    }

    // ================= FC stack (BN over batch axis, M=32) =======================
    float* latent = (float*)d_out + (size_t)B * NPTS * 3;   // recon first, then latent
    for (int i = 9; i < 13; ++i) {
        run_gemm(stream, buf1, Wt[i], bias_of(i), buf0, nullptr, B, LY[i].N, LY[i].Kpad);
        run_bn(stream, buf0, gamma_of(i), beta_of(i), meanb, invb, buf1,
               (i == 10) ? latent : nullptr, B, LY[i].N);   // latent after fc layer 1
    }

    // ================= Output: (32,1024) x (1024,12288) + bo -> recon ============
    run_gemm(stream, buf1, Wt[13], P[53], nullptr, (float*)d_out, B, 12288, 1024);
}
